// FourRegionAttention_66571993089021
// MI455X (gfx1250) — matrix operand order repaired and checked by reading, not compiled, not run on hardware
//
#include <hip/hip_runtime.h>
#include <hip/hip_bf16.h>

typedef __bf16 bf16_t;
typedef __attribute__((ext_vector_type(16))) __bf16 v16bf;
typedef __attribute__((ext_vector_type(8)))  __bf16 v8bf;
typedef __attribute__((ext_vector_type(8)))  float  v8f;

#define NB 64
#define NTOK 2304        // 48*48
#define NHW 48
#define NDIM 64
#define NHEADS 8
#define NROWS (NB*NTOK)  // 147456

#define USE_ASYNC_LDS 1

static __device__ __forceinline__ v8f wmma_bf16(v16bf a, v16bf b, v8f c) {
  return __builtin_amdgcn_wmma_f32_16x16x32_bf16(false, a, false, b, (short)0, c, false, false);
}

// A fragment (16x32 bf16, M x K), contiguous K order (matches B):
//   lanes 0-15: M=lane, elems 0..15 -> K=kbase+0..15
//   lanes16-31: M=lane-16, elems 0..15 -> K=kbase+16..31
static __device__ __forceinline__ v16bf make_afrag(const bf16_t* rowPtr, int kbase, int hs) {
  v8bf p0 = *(const v8bf*)(rowPtr + kbase + hs * 16);
  v8bf p1 = *(const v8bf*)(rowPtr + kbase + hs * 16 + 8);
  return __builtin_shufflevector(p0, p1, 0,1,2,3,4,5,6,7,8,9,10,11,12,13,14,15);
}

// B fragment (32x16 bf16, K x N), W stored row-major (N,64):
//   lane holds column N=lane&15; K = (lane<16?0:16)+e contiguous
static __device__ __forceinline__ v16bf make_bfrag(const bf16_t* W, int col, int kbase, int hs) {
  return *(const v16bf*)(W + (size_t)col * 64 + kbase + hs * 16);
}

// Block-cooperative stage of `bytes` (multiple of 16, all threads active) from
// global to LDS. Uses gfx1250 GLOBAL_LOAD_ASYNC_TO_LDS_B128 (ASYNCcnt-tracked,
// no VGPR round trip), then s_wait_asynccnt + workgroup barrier.
static __device__ __forceinline__ void async_stage_to_lds(bf16_t* ldsDst,
                                                          const bf16_t* gSrc,
                                                          int bytes) {
#if USE_ASYNC_LDS
  typedef __attribute__((address_space(3))) bf16_t* lds_ptr_t;
  unsigned ldsBase = (unsigned)(unsigned long long)(lds_ptr_t)ldsDst;
  const char* g = (const char*)gSrc;
  for (int off = threadIdx.x * 16; off < bytes; off += blockDim.x * 16) {
    unsigned l = ldsBase + (unsigned)off;
    unsigned long long ga = (unsigned long long)(g + off);
    asm volatile("global_load_async_to_lds_b128 %0, %1, off"
                 :: "v"(l), "v"(ga) : "memory");
  }
  asm volatile("s_wait_asynccnt 0x0" ::: "memory");
  __syncthreads();
#else
  uint4* d = (uint4*)ldsDst;
  const uint4* s = (const uint4*)gSrc;
  int n16 = bytes >> 4;
  for (int i = threadIdx.x; i < n16; i += blockDim.x) d[i] = s[i];
  __syncthreads();
#endif
}

// ---------------- K0: weight fusion precompute (1 block) ----------------
__global__ __launch_bounds__(256) void precompute_kernel(
    const float* __restrict__ Wqkv, const float* __restrict__ bqkv,
    const float* __restrict__ cls_tok,
    const float* __restrict__ Wwqkv, const float* __restrict__ bwqkv,
    const float* __restrict__ Wwproj, const float* __restrict__ bwproj,
    const float* __restrict__ Wraq, const float* __restrict__ Wrak,
    const float* __restrict__ Wrav, const float* __restrict__ Wrao,
    const float* __restrict__ brao, const float* __restrict__ Wout,
    const float* __restrict__ bout,
    float* qc, float* vc, float* clsl, float* qpb, float* bkv2, float* bfin,
    float* bKV, bf16_t* WcombQ, bf16_t* Wkv2, bf16_t* Whov, bf16_t* WKV) {
  __shared__ float ctmp[192];
  const int tid = threadIdx.x;
  const float sw = 0.3535533905932738f;  // 1/sqrt(8)

  for (int i = tid; i < 192; i += 256) {
    float s = bwqkv[i];
    for (int d = 0; d < 64; ++d) s += Wwqkv[i * 64 + d] * cls_tok[d];
    ctmp[i] = s;
  }
  __syncthreads();
  if (tid < 64) { qc[tid] = ctmp[tid] * sw; vc[tid] = ctmp[128 + tid]; }
  if (tid < 8) {
    float s = 0.f;
    for (int d = 0; d < 8; ++d) s += ctmp[tid * 8 + d] * sw * ctmp[64 + tid * 8 + d];
    clsl[tid] = s;
  }
  // WcombQ[h*64+e][d] = sum_t Wraq[e][t] * Wqkv[h*64+t][d]
  for (int idx = tid; idx < 512 * 64; idx += 256) {
    int row = idx >> 6, d = idx & 63, h = row >> 6, e = row & 63;
    float s = 0.f;
    for (int t = 0; t < 64; ++t) s += Wraq[e * 64 + t] * Wqkv[(h * 64 + t) * 64 + d];
    WcombQ[idx] = (bf16_t)s;
  }
  for (int idx = tid; idx < 512; idx += 256) {
    int h = idx >> 6, e = idx & 63;
    float s = 0.f;
    for (int t = 0; t < 64; ++t) s += Wraq[e * 64 + t] * bqkv[h * 64 + t];
    qpb[idx] = s;
  }
  // Wkv2[j][d] = sum_t {Wrak|Wrav}[j][t] * Wwproj[t][d]
  for (int idx = tid; idx < 128 * 64; idx += 256) {
    int j = idx >> 6, d = idx & 63;
    const float* wr = (j < 64) ? (Wrak + j * 64) : (Wrav + (j - 64) * 64);
    float s = 0.f;
    for (int t = 0; t < 64; ++t) s += wr[t] * Wwproj[t * 64 + d];
    Wkv2[idx] = (bf16_t)s;
  }
  for (int j = tid; j < 128; j += 256) {
    const float* wr = (j < 64) ? (Wrak + j * 64) : (Wrav + (j - 64) * 64);
    float s = 0.f;
    for (int t = 0; t < 64; ++t) s += wr[t] * bwproj[t];
    bkv2[j] = s;
  }
  // Whov[(h*64+i)*64+d] = sum_j Wout[i][h*64+j] * Wrao[j][d]
  for (int idx = tid; idx < 8 * 64 * 64; idx += 256) {
    int h = idx >> 12, rem = idx & 4095, i = rem >> 6, d = rem & 63;
    float s = 0.f;
    for (int j = 0; j < 64; ++j) s += Wout[i * 512 + h * 64 + j] * Wrao[j * 64 + d];
    Whov[idx] = (bf16_t)s;
  }
  for (int i = tid; i < 64; i += 256) {
    float s = bout[i];
    for (int c = 0; c < 512; ++c) s += Wout[i * 512 + c] * brao[c & 63];
    bfin[i] = s;
  }
  for (int idx = tid; idx < 128 * 64; idx += 256) WKV[idx] = (bf16_t)Wwqkv[4096 + idx];
  for (int j = tid; j < 128; j += 256) bKV[j] = bwqkv[64 + j];
}

// ---------------- K1: LayerNorm -> bf16, one wave per row ----------------
__global__ __launch_bounds__(256) void ln_kernel(const float* __restrict__ X,
                                                 const float* __restrict__ g,
                                                 const float* __restrict__ bl,
                                                 bf16_t* __restrict__ XN) {
  const int lane = threadIdx.x & 31, wave = threadIdx.x >> 5;
  const long row = (long)blockIdx.x * 8 + wave;
  const float* xr = X + row * 64;
  float a = xr[lane], b = xr[lane + 32];
  float s = a + b;
  #pragma unroll
  for (int off = 16; off; off >>= 1) s += __shfl_xor(s, off, 32);
  float mu = s * (1.f / 64.f);
  float da = a - mu, db = b - mu;
  float v = da * da + db * db;
  #pragma unroll
  for (int off = 16; off; off >>= 1) v += __shfl_xor(v, off, 32);
  float rinv = rsqrtf(v * (1.f / 64.f) + 1e-5f);
  XN[row * 64 + lane]      = (bf16_t)(da * rinv * g[lane] + bl[lane]);
  XN[row * 64 + lane + 32] = (bf16_t)(db * rinv * g[lane + 32] + bl[lane + 32]);
}

// ---------------- K2: WMMA GEMM (N=128) kv + fused logits epilogue ----------------
__global__ __launch_bounds__(256) void gemm_kv_logits(
    const bf16_t* __restrict__ A, const bf16_t* __restrict__ W,
    const float* __restrict__ bias, const float* __restrict__ qc,
    bf16_t* __restrict__ Vout, float* __restrict__ Logits) {
  __shared__ bf16_t sW[128 * 64];   // 16 KB staged weights
  __shared__ float tile[8][16][128];
  async_stage_to_lds(sW, W, 128 * 64 * 2);

  const int lane = threadIdx.x & 31, wave = threadIdx.x >> 5;
  const long rowBase = (long)blockIdx.x * 128;
  const long waveRow = rowBase + wave * 16;
  const int hs = lane >> 4;
  const bf16_t* arow = A + (waveRow + (lane & 15)) * 64;
  v16bf a0 = make_afrag(arow, 0, hs);
  v16bf a1 = make_afrag(arow, 32, hs);
  #pragma unroll
  for (int j = 0; j < 8; ++j) {
    int col = j * 16 + (lane & 15);
    v16bf b0 = make_bfrag(sW, col, 0, hs);
    v16bf b1 = make_bfrag(sW, col, 32, hs);
    v8f acc = {};
    acc = wmma_bf16(a0, b0, acc);
    acc = wmma_bf16(a1, b1, acc);
    float cb = bias[col];
    #pragma unroll
    for (int r = 0; r < 8; ++r) tile[wave][r + hs * 8][col] = acc[r] + cb;
  }
  __syncthreads();
  // v-part (cols 64..127) -> bf16 rows (B*n, 64)
  for (int idx = threadIdx.x; idx < 128 * 64; idx += 256) {
    int row = idx >> 6, c = idx & 63;
    Vout[(rowBase + row) * 64 + c] = (bf16_t)tile[row >> 4][row & 15][64 + c];
  }
  // logits[b,h,n] = q_c[h] . k[h]
  for (int idx = threadIdx.x; idx < 128 * 8; idx += 256) {
    int row = idx >> 3, head = idx & 7;
    float s = 0.f;
    #pragma unroll
    for (int d = 0; d < 8; ++d) s += qc[head * 8 + d] * tile[row >> 4][row & 15][head * 8 + d];
    long g = rowBase + row;
    long b = g / NTOK, nn = g - b * NTOK;
    Logits[(b * 8 + head) * NTOK + nn] = s;
  }
}

// ---------------- K4: per-(b,h) quadrant prefix sums ----------------
static __device__ __forceinline__ float prefv(const float* sC, int i, int j, int ch) {
  return (i < 0 || j < 0) ? 0.f : sC[(i * 48 + j) * 9 + ch];
}

__global__ __launch_bounds__(256) void quad_kernel(
    const float* __restrict__ Logits, const bf16_t* __restrict__ Vbf,
    const float* __restrict__ vc, const float* __restrict__ clsl,
    bf16_t* __restrict__ ClsOut) {
  __shared__ float sC[48 * 48 * 9];
  __shared__ float sRed[256];
  const int tid = threadIdx.x;
  const int b = blockIdx.x >> 3, h = blockIdx.x & 7;
  const float* lrow = Logits + (size_t)blockIdx.x * NTOK;

  float lm = -1e30f;
  for (int p = tid; p < NTOK; p += 256) lm = fmaxf(lm, lrow[p]);
  sRed[tid] = lm;
  __syncthreads();
  for (int s = 128; s > 0; s >>= 1) {
    if (tid < s) sRed[tid] = fmaxf(sRed[tid], sRed[tid + s]);
    __syncthreads();
  }
  const float cl = clsl[h];
  const float m = fmaxf(sRed[0], cl);
  const float ecls = __expf(cl - m);

  for (int p = tid; p < NTOK; p += 256) {
    float e = __expf(lrow[p] - m);
    float* dst = sC + p * 9;
    dst[0] = e;
    v8bf vv = *(const v8bf*)(Vbf + ((size_t)b * NTOK + p) * 64 + h * 8);
    #pragma unroll
    for (int d = 0; d < 8; ++d) dst[1 + d] = e * (float)vv[d];
  }
  __syncthreads();
  // cumsum along W
  for (int l = tid; l < 48 * 9; l += 256) {
    int row = l / 9, ch = l % 9;
    float run = sC[(row * 48) * 9 + ch];
    for (int w = 1; w < 48; ++w) {
      int idx = (row * 48 + w) * 9 + ch;
      run += sC[idx];
      sC[idx] = run;
    }
  }
  __syncthreads();
  // cumsum along H
  for (int l = tid; l < 48 * 9; l += 256) {
    int cw = l / 9, ch = l % 9;
    float run = sC[cw * 9 + ch];
    for (int hh = 1; hh < 48; ++hh) {
      int idx = (hh * 48 + cw) * 9 + ch;
      run += sC[idx];
      sC[idx] = run;
    }
  }
  __syncthreads();
  for (int idx = tid; idx < NTOK * 4; idx += 256) {
    int p = idx >> 2, r = idx & 3;
    int hh = p / 48, ww = p % 48;
    float S[9];
    switch (r) {
      case 0:
        #pragma unroll
        for (int ch = 0; ch < 9; ++ch) S[ch] = prefv(sC, hh, ww, ch);
        break;
      case 1:
        #pragma unroll
        for (int ch = 0; ch < 9; ++ch) S[ch] = prefv(sC, hh, 47, ch) - prefv(sC, hh, ww - 1, ch);
        break;
      case 2:
        #pragma unroll
        for (int ch = 0; ch < 9; ++ch) S[ch] = prefv(sC, 47, ww, ch) - prefv(sC, hh - 1, ww, ch);
        break;
      default:
        #pragma unroll
        for (int ch = 0; ch < 9; ++ch)
          S[ch] = prefv(sC, 47, 47, ch) - prefv(sC, hh - 1, 47, ch) -
                  prefv(sC, 47, ww - 1, ch) + prefv(sC, hh - 1, ww - 1, ch);
        break;
    }
    float inv = 1.0f / (ecls + S[0]);
    bf16_t* outp = ClsOut + (((size_t)b * NTOK + p) * 4 + r) * 64 + h * 8;
    #pragma unroll
    for (int d = 0; d < 8; ++d) outp[d] = (bf16_t)((ecls * vc[h * 8 + d] + S[1 + d]) * inv);
  }
}

// ---------------- generic WMMA GEMM: Out(M,N) = A(M,64) * W(N,64)^T + bias ----------------
// Weights staged to LDS once per block (async), shared by all 8 waves.
__global__ __launch_bounds__(256) void gemm_bf16_generic(
    const bf16_t* __restrict__ A, const bf16_t* __restrict__ W,
    const float* __restrict__ bias, bf16_t* __restrict__ Out, int N) {
  __shared__ bf16_t sW[512 * 64];  // up to 64 KB staged weights
  async_stage_to_lds(sW, W, N * 64 * 2);

  const int lane = threadIdx.x & 31, wave = threadIdx.x >> 5;
  const long rowBase = (long)blockIdx.x * 128 + wave * 16;
  const int hs = lane >> 4;
  const bf16_t* arow = A + (rowBase + (lane & 15)) * 64;
  v16bf a0 = make_afrag(arow, 0, hs);
  v16bf a1 = make_afrag(arow, 32, hs);
  for (int n0 = 0; n0 < N; n0 += 16) {
    int col = n0 + (lane & 15);
    v16bf b0 = make_bfrag(sW, col, 0, hs);
    v16bf b1 = make_bfrag(sW, col, 32, hs);
    v8f acc = {};
    acc = wmma_bf16(a0, b0, acc);
    acc = wmma_bf16(a1, b1, acc);
    float cb = bias[col];
    #pragma unroll
    for (int r = 0; r < 8; ++r) {
      long mrow = rowBase + r + hs * 8;
      Out[mrow * N + col] = (bf16_t)(acc[r] + cb);
    }
  }
}

// ---------------- K7: scores/softmax/agg + WMMA output contraction ----------------
__global__ __launch_bounds__(128) void final_kernel(
    const bf16_t* __restrict__ QP, const bf16_t* __restrict__ KPVP,
    const bf16_t* __restrict__ Whov, const float* __restrict__ bfin,
    float* __restrict__ Out) {
  __shared__ bf16_t sWh[512 * 64];  // 64 KB staged Whov
  __shared__ bf16_t sAgg[4][16][512];
  async_stage_to_lds(sWh, Whov, 512 * 64 * 2);

  const int lane = threadIdx.x & 31, wave = threadIdx.x >> 5;
  const long posBase = (long)blockIdx.x * 64 + wave * 16;
  const float scale_ra = 0.3535533905932738f;

  // phase 1: each of 4 iterations handles one (pos,head) pair per lane
  for (int i = 0; i < 4; ++i) {
    int pair = i * 32 + lane;
    int p = pair >> 3, head = pair & 7;
    long gp = posBase + p;
    float qv[64];
    const v8bf* qrow = (const v8bf*)(QP + (gp * 8 + head) * 64);
    #pragma unroll
    for (int c = 0; c < 8; ++c) {
      v8bf t = qrow[c];
      #pragma unroll
      for (int k = 0; k < 8; ++k) qv[c * 8 + k] = (float)t[k];
    }
    float sc[4];
    #pragma unroll
    for (int r = 0; r < 4; ++r) {
      const v8bf* krow = (const v8bf*)(KPVP + (gp * 4 + r) * 128);
      float s = 0.f;
      #pragma unroll
      for (int c = 0; c < 8; ++c) {
        v8bf t = krow[c];
        #pragma unroll
        for (int k = 0; k < 8; ++k) s += qv[c * 8 + k] * (float)t[k];
      }
      sc[r] = s * scale_ra;
    }
    float mx = fmaxf(fmaxf(sc[0], sc[1]), fmaxf(sc[2], sc[3]));
    float ex[4], den = 0.f;
    #pragma unroll
    for (int r = 0; r < 4; ++r) { ex[r] = __expf(sc[r] - mx); den += ex[r]; }
    float dinv = 1.0f / den;
    float agg[64];
    #pragma unroll
    for (int d = 0; d < 64; ++d) agg[d] = 0.f;
    #pragma unroll
    for (int r = 0; r < 4; ++r) {
      const v8bf* vrow = (const v8bf*)(KPVP + (gp * 4 + r) * 128 + 64);
      float ar = ex[r] * dinv;
      #pragma unroll
      for (int c = 0; c < 8; ++c) {
        v8bf t = vrow[c];
        #pragma unroll
        for (int k = 0; k < 8; ++k) agg[c * 8 + k] += ar * (float)t[k];
      }
    }
    #pragma unroll
    for (int d = 0; d < 64; ++d) sAgg[wave][p][head * 64 + d] = (bf16_t)agg[d];
  }
  __syncthreads();

  // phase 2: Out(16,64) = Agg(16,512) * Whov(64,512)^T  via 4x16 WMMAs
  const int hs = lane >> 4;
  const bf16_t* aggRow = &sAgg[wave][lane & 15][0];
  for (int nt = 0; nt < 4; ++nt) {
    int col = nt * 16 + (lane & 15);
    v8f acc = {};
    #pragma unroll
    for (int kb = 0; kb < 16; ++kb) {
      int k0 = kb * 32;
      v16bf a = make_afrag(aggRow, k0, hs);
      int head = k0 >> 6, db = k0 & 63;
      v16bf bv = *(const v16bf*)(sWh + ((size_t)(head * 64 + col)) * 64 + db + hs * 16);
      acc = wmma_bf16(a, bv, acc);
    }
    float cb = bfin[col];
    #pragma unroll
    for (int r = 0; r < 8; ++r) {
      long mrow = posBase + r + hs * 8;
      Out[mrow * 64 + col] = acc[r] + cb;
    }
  }
}

extern "C" void kernel_launch(void* const* d_in, const int* in_sizes, int n_in,
                              void* d_out, int out_size, void* d_ws, size_t ws_size,
                              hipStream_t stream) {
  const float* x       = (const float*)d_in[0];
  const float* ln_g    = (const float*)d_in[1];
  const float* ln_b    = (const float*)d_in[2];
  const float* Wqkv    = (const float*)d_in[3];
  const float* bqkv    = (const float*)d_in[4];
  const float* cls_tok = (const float*)d_in[5];
  const float* Wwqkv   = (const float*)d_in[6];
  const float* bwqkv   = (const float*)d_in[7];
  const float* Wwproj  = (const float*)d_in[8];
  const float* bwproj  = (const float*)d_in[9];
  const float* Wraq    = (const float*)d_in[10];
  const float* Wrak    = (const float*)d_in[11];
  const float* Wrav    = (const float*)d_in[12];
  const float* Wrao    = (const float*)d_in[13];
  const float* brao    = (const float*)d_in[14];
  const float* Wout    = (const float*)d_in[15];
  const float* bout    = (const float*)d_in[16];
  float* out = (float*)d_out;

  char* ws = (char*)d_ws;
  size_t off = 0;
  auto nxt = [&](size_t bytes) {
    size_t r = off;
    off += (bytes + 255) & ~(size_t)255;
    return r;
  };
  size_t o_qc     = nxt(64 * 4);
  size_t o_vc     = nxt(64 * 4);
  size_t o_clsl   = nxt(8 * 4);
  size_t o_qpb    = nxt(512 * 4);
  size_t o_bkv2   = nxt(128 * 4);
  size_t o_bfin   = nxt(64 * 4);
  size_t o_bKV    = nxt(128 * 4);
  size_t o_WcombQ = nxt(512 * 64 * 2);
  size_t o_Wkv2   = nxt(128 * 64 * 2);
  size_t o_Whov   = nxt(512 * 64 * 2);
  size_t o_WKV    = nxt(128 * 64 * 2);
  size_t o_xn     = nxt((size_t)NROWS * 64 * 2);
  size_t o_vbf    = nxt((size_t)NROWS * 64 * 2);
  size_t o_logits = nxt((size_t)NB * 8 * NTOK * 4);
  size_t o_clsout = nxt((size_t)NROWS * 4 * 64 * 2);
  size_t o_kpvp   = nxt((size_t)NROWS * 4 * 128 * 2);
  size_t o_qp     = nxt((size_t)NROWS * 512 * 2);

  float*  qc     = (float*)(ws + o_qc);
  float*  vc     = (float*)(ws + o_vc);
  float*  clsl   = (float*)(ws + o_clsl);
  float*  qpb    = (float*)(ws + o_qpb);
  float*  bkv2   = (float*)(ws + o_bkv2);
  float*  bfin   = (float*)(ws + o_bfin);
  float*  bKV    = (float*)(ws + o_bKV);
  bf16_t* WcombQ = (bf16_t*)(ws + o_WcombQ);
  bf16_t* Wkv2   = (bf16_t*)(ws + o_Wkv2);
  bf16_t* Whov   = (bf16_t*)(ws + o_Whov);
  bf16_t* WKV    = (bf16_t*)(ws + o_WKV);
  bf16_t* xn     = (bf16_t*)(ws + o_xn);
  bf16_t* vbf    = (bf16_t*)(ws + o_vbf);
  float*  logits = (float*)(ws + o_logits);
  bf16_t* clsout = (bf16_t*)(ws + o_clsout);
  bf16_t* kpvp   = (bf16_t*)(ws + o_kpvp);
  bf16_t* qp     = (bf16_t*)(ws + o_qp);

  precompute_kernel<<<1, 256, 0, stream>>>(
      Wqkv, bqkv, cls_tok, Wwqkv, bwqkv, Wwproj, bwproj, Wraq, Wrak, Wrav,
      Wrao, brao, Wout, bout, qc, vc, clsl, qpb, bkv2, bfin, bKV, WcombQ,
      Wkv2, Whov, WKV);

  ln_kernel<<<NROWS / 8, 256, 0, stream>>>(x, ln_g, ln_b, xn);

  gemm_kv_logits<<<NROWS / 128, 256, 0, stream>>>(xn, WKV, bKV, qc, vbf, logits);

  quad_kernel<<<NB * NHEADS, 256, 0, stream>>>(logits, vbf, vc, clsl, clsout);

  gemm_bf16_generic<<<(NROWS * 4) / 128, 256, 0, stream>>>(clsout, Wkv2, bkv2, kpvp, 128);

  gemm_bf16_generic<<<NROWS / 128, 256, 0, stream>>>(xn, WcombQ, qpb, qp, 512);

  final_kernel<<<NROWS / 64, 128, 0, stream>>>(qp, kpvp, Whov, bfin, out);

  (void)in_sizes; (void)n_in; (void)out_size; (void)ws_size;
}
